// SeqAttention_36275293782059
// MI455X (gfx1250) — matrix operand order repaired and checked by reading, not compiled, not run on hardware
//
#include <hip/hip_runtime.h>
#include <hip/hip_fp16.h>

#define C_IN   512
#define C_HID  32
#define NH     16
#define CZ     128
#define BB     2
#define NN     1024
#define HD     512              // NH*C_HID
#define MTOT   (BB*NN)          // 2048
#define INF_   1e9f
#define EPS_   1e-5f

typedef __attribute__((ext_vector_type(16))) _Float16 v16h;
typedef __attribute__((ext_vector_type(8)))  _Float16 v8h;
typedef __attribute__((ext_vector_type(4)))  _Float16 v4h;
typedef __attribute__((ext_vector_type(8)))  float    v8f;

union V16U { v16h v; v8h h[2]; };

static __device__ inline v8f vzero8() {
  v8f v = {0.f, 0.f, 0.f, 0.f, 0.f, 0.f, 0.f, 0.f};
  return v;
}
static __device__ inline float sum4(float4 a)   { return a.x + a.y + a.z + a.w; }
static __device__ inline float sumsq4(float4 a) {
  return a.x * a.x + a.y * a.y + a.z * a.z + a.w * a.w;
}
static __device__ inline v8h cvt8(float4 a, float4 b) {
  v8h r;
  r[0] = (_Float16)a.x; r[1] = (_Float16)a.y;
  r[2] = (_Float16)a.z; r[3] = (_Float16)a.w;
  r[4] = (_Float16)b.x; r[5] = (_Float16)b.y;
  r[6] = (_Float16)b.z; r[7] = (_Float16)b.w;
  return r;
}

// ---------------------------------------------------------------------------
// K0: weight convert + transpose  W[k][n] (f32) -> Wt[n][k] (f16)
// ---------------------------------------------------------------------------
__global__ void convert_w_kernel(const float* __restrict__ W,
                                 _Float16* __restrict__ Wt, int K, int N) {
  int idx = blockIdx.x * blockDim.x + threadIdx.x;
  if (idx >= K * N) return;
  int k = idx / N, n = idx % N;
  Wt[(size_t)n * K + k] = (_Float16)W[idx];
}

// ---------------------------------------------------------------------------
// K0b: fold LN(z) gamma into w_z:  Wzgt[h][c] = f16(gamma[c]*wz[c][h])
//      SC[h]    = sum_c gamma[c]*wz[c][h]   (S_h)
//      SC[16+h] = sum_c beta[c] *wz[c][h]   (C_h)
// ---------------------------------------------------------------------------
__global__ void prep_wz_kernel(const float* __restrict__ wz,
                               const float* __restrict__ lnw,
                               const float* __restrict__ lnb,
                               _Float16* __restrict__ Wzgt,
                               float* __restrict__ SC) {
  int i = blockIdx.x * blockDim.x + threadIdx.x;
  if (i < CZ * NH) {
    int c = i / NH, h = i % NH;
    Wzgt[h * CZ + c] = (_Float16)(lnw[c] * wz[i]);
  }
  if (i < 2 * NH) {
    int h = i & 15;
    const float* g = (i < NH) ? lnw : lnb;
    float acc = 0.f;
    for (int c = 0; c < CZ; ++c) acc += g[c] * wz[c * NH + h];
    SC[i] = acc;
  }
}

// ---------------------------------------------------------------------------
// K1: LayerNorm of m -> f16  (one wave per 512-wide row, 8 waves/block)
// ---------------------------------------------------------------------------
__global__ void ln_m_kernel(const float* __restrict__ m,
                            const float* __restrict__ w,
                            const float* __restrict__ b,
                            _Float16* __restrict__ xbf) {
  int row  = blockIdx.x * 8 + (threadIdx.x >> 5);
  int lane = threadIdx.x & 31;
  const float* mr = m + (size_t)row * C_IN;
  float4 vals[4];
  float s = 0.f, s2 = 0.f;
#pragma unroll
  for (int i = 0; i < 4; ++i) {
    vals[i] = ((const float4*)mr)[lane + 32 * i];
    s  += sum4(vals[i]);
    s2 += sumsq4(vals[i]);
  }
#pragma unroll
  for (int msk = 16; msk >= 1; msk >>= 1) {
    s  += __shfl_xor(s, msk, 32);
    s2 += __shfl_xor(s2, msk, 32);
  }
  float mu   = s * (1.f / C_IN);
  float var  = s2 * (1.f / C_IN) - mu * mu;
  float rinv = rsqrtf(var + EPS_);
  _Float16* xr = xbf + (size_t)row * C_IN;
#pragma unroll
  for (int i = 0; i < 4; ++i) {
    int c4 = lane + 32 * i;
    float4 wv = ((const float4*)w)[c4];
    float4 bv = ((const float4*)b)[c4];
    v4h o;
    o[0] = (_Float16)((vals[i].x - mu) * rinv * wv.x + bv.x);
    o[1] = (_Float16)((vals[i].y - mu) * rinv * wv.y + bv.y);
    o[2] = (_Float16)((vals[i].z - mu) * rinv * wv.z + bv.z);
    o[3] = (_Float16)((vals[i].w - mu) * rinv * wv.w + bv.w);
    *(v4h*)(xr + c4 * 4) = o;
  }
}

// ---------------------------------------------------------------------------
// K2: WMMA GEMM  [2048,512]f16 x Wt[512(n-major),512] -> 16x16 tile per wave
// mode 0/1/2: Q/K/V -> f16 [b,h,n,d]; mode 3: G=sigmoid(x+b_g) f32;
// mode 4: final out = x + b_o f32
// ---------------------------------------------------------------------------
__global__ void gemm_wmma_kernel(const _Float16* __restrict__ X,
                                 const _Float16* __restrict__ Wt,
                                 _Float16* __restrict__ outH,
                                 float* __restrict__ outF,
                                 const float* __restrict__ bvec, int mode) {
  int lane = threadIdx.x & 31;
  int tile = blockIdx.x * 8 + (threadIdx.x >> 5);
  int tm = tile >> 5;                 // 128 row tiles
  int tn = tile & 31;                 // 32 col tiles
  int hi = lane >> 4, l15 = lane & 15;
  int row  = tm * 16 + l15;
  int coln = tn * 16 + l15;
  const _Float16* xr = X  + (size_t)row  * C_IN;
  const _Float16* wr = Wt + (size_t)coln * C_IN;
  int kb = hi * 16;
  v8f acc = vzero8();
#pragma unroll 4
  for (int k0 = 0; k0 < C_IN; k0 += 32) {
    V16U a;
    a.h[0] = *(const v8h*)(xr + k0 + kb);
    a.h[1] = *(const v8h*)(xr + k0 + kb + 8);
    v16h bf = *(const v16h*)(wr + k0 + hi * 16);
    acc = __builtin_amdgcn_wmma_f32_16x16x32_f16(false, a.v, false, bf,
                                                 (short)0, acc, false, false);
  }
#pragma unroll
  for (int r = 0; r < 8; ++r) {
    int grow = tm * 16 + r + 8 * hi;        // 0..2047
    int col  = coln;
    if (mode <= 2) {
      float v = acc[r] * (mode == 0 ? 0.17677669529663689f : 1.f);
      int bb = grow >> 10, nseq = grow & 1023;
      int h = col >> 5, d = col & 31;
      outH[((size_t)(bb * NH + h) * NN + nseq) * C_HID + d] = (_Float16)v;
    } else if (mode == 3) {
      float v = acc[r] + bvec[col];
      outF[(size_t)grow * HD + col] = 1.f / (1.f + __expf(-v));
    } else {
      outF[(size_t)grow * HD + col] = acc[r] + bvec[col];
    }
  }
}

// ---------------------------------------------------------------------------
// K3: pair bias via WMMA. One wave = 16 consecutive (b,q,k) pairs (an M-tile).
// bias = rinv*(z . (gamma*wz) - mu*S_h) + C_h  -- LN folded into the weight.
// A = z tile [16 pairs x 128ch] f16 (4 K-slices), B = Wzgt, C-frag epilogue
// applies per-row (mu, rinv) broadcast by lane shuffles. ~18 instrs/pair ->
// HBM-bound at 23.3 TB/s over the 1 GiB z tensor.
// ---------------------------------------------------------------------------
__global__ void pair_bias_wmma_kernel(const float* __restrict__ z,
                                      const _Float16* __restrict__ Wzgt,
                                      const float* __restrict__ SC,
                                      float* __restrict__ bias) {
  int wave = threadIdx.x >> 5, lane = threadIdx.x & 31;
  int hi = lane >> 4, l15 = lane & 15, kb = hi * 16;
  size_t tile = (size_t)blockIdx.x * 8 + wave;   // 131072 tiles of 16 pairs
  size_t p0 = tile * 16;
  const float* zr = z + (p0 + l15) * CZ;

  float Sh = SC[l15];
  float Ch = SC[16 + l15];

  v16h bw[4];
#pragma unroll
  for (int f = 0; f < 4; ++f)
    bw[f] = *(const v16h*)(Wzgt + l15 * CZ + f * 32 + hi * 16);

  float s = 0.f, s2 = 0.f;
  V16U afrag[4];
#pragma unroll
  for (int f = 0; f < 4; ++f) {
    const float* cp = zr + f * 32 + kb;
    float4 a0 = *(const float4*)(cp);
    float4 a1 = *(const float4*)(cp + 4);
    float4 a2 = *(const float4*)(cp + 8);
    float4 a3 = *(const float4*)(cp + 12);
    s  += sum4(a0) + sum4(a1) + sum4(a2) + sum4(a3);
    s2 += sumsq4(a0) + sumsq4(a1) + sumsq4(a2) + sumsq4(a3);
    afrag[f].h[0] = cvt8(a0, a1);
    afrag[f].h[1] = cvt8(a2, a3);
  }
  // lane pair (L, L^16) together hold the full 128-channel row
  s  += __shfl_xor(s, 16, 32);
  s2 += __shfl_xor(s2, 16, 32);
  float mu   = s * (1.f / CZ);
  float var  = s2 * (1.f / CZ) - mu * mu;
  float rinv = rsqrtf(var + EPS_);

  v8f acc = vzero8();
#pragma unroll
  for (int f = 0; f < 4; ++f)
    acc = __builtin_amdgcn_wmma_f32_16x16x32_f16(false, afrag[f].v, false,
                                                 bw[f], (short)0, acc,
                                                 false, false);

  int k0 = (int)(p0 & (NN - 1));
  int qi = (int)((p0 >> 10) & (NN - 1));
  int bb = (int)(p0 >> 20);
  float* op = bias + (((size_t)(bb * NH + l15) * NN + qi) * NN) + k0 + 8 * hi;
#pragma unroll
  for (int r = 0; r < 8; ++r) {
    int src = r + 8 * hi;                  // lane 'src' holds row src's stats
    float mu_r   = __shfl(mu, src, 32);
    float rinv_r = __shfl(rinv, src, 32);
    op[r] = rinv_r * (acc[r] - mu_r * Sh) + Ch;
  }
}

// ---------------------------------------------------------------------------
// K4: flash attention with pair bias. Block = 8 waves = 128 queries of one
// (b,h). K staged via global_load_async_to_lds_b128 (ASYNCcnt), V^T staged
// manually; online softmax in C-fragment layout; P through per-wave LDS tile.
// ---------------------------------------------------------------------------
__global__ void attn_kernel(const _Float16* __restrict__ Q,
                            const _Float16* __restrict__ Kb,
                            const _Float16* __restrict__ Vb,
                            const float* __restrict__ bias,
                            const float* __restrict__ mask,
                            float* __restrict__ O) {
  extern __shared__ char smem[];
  _Float16* Ks = (_Float16*)smem;                     // [1024][32] @ lds 0
  _Float16* Vt = (_Float16*)(smem + 65536);           // [32][1024] (transposed)
  _Float16* Pt = (_Float16*)(smem + 131072);          // 8 x [16][32]

  int wave = threadIdx.x >> 5, lane = threadIdx.x & 31;
  int hi = lane >> 4, l15 = lane & 15;
  int bh = blockIdx.x >> 3;
  int qc = blockIdx.x & 7;
  int b = bh >> 4, h = bh & 15;

  const _Float16* Kg = Kb + (size_t)bh * NN * C_HID;
  const _Float16* Vg = Vb + (size_t)bh * NN * C_HID;

  // --- async stage: K rows -> LDS (16B per lane per issue, ASYNCcnt) ---
  for (int i = threadIdx.x; i < NN * C_HID / 8; i += 256) {
    unsigned ldsOff = (unsigned)(i * 16);             // Ks sits at LDS base
    const _Float16* gp = Kg + (size_t)i * 8;
    asm volatile("global_load_async_to_lds_b128 %0, %1, off"
                 :: "v"(ldsOff), "v"(gp) : "memory");
  }
  // --- manual transposed stage of V ---
  for (int i = threadIdx.x; i < NN * C_HID / 4; i += 256) {
    int key = i >> 3;
    int d0  = (i & 7) * 4;
    const _Float16* vp = Vg + (size_t)key * C_HID + d0;
    Vt[(d0 + 0) * NN + key] = vp[0];
    Vt[(d0 + 1) * NN + key] = vp[1];
    Vt[(d0 + 2) * NN + key] = vp[2];
    Vt[(d0 + 3) * NN + key] = vp[3];
  }
  asm volatile("s_wait_asynccnt 0x0" ::: "memory");
  __syncthreads();

  int q0 = qc * 128 + wave * 16;
  int kb = hi * 16;
  const _Float16* qp = Q + ((size_t)bh * NN + q0 + l15) * C_HID;
  V16U aq;
  aq.h[0] = *(const v8h*)(qp + kb);
  aq.h[1] = *(const v8h*)(qp + kb + 8);

  v8f o0 = vzero8(), o1 = vzero8();
  float rmax[8], rsum[8];
#pragma unroll
  for (int r = 0; r < 8; ++r) { rmax[r] = -1e30f; rsum[r] = 0.f; }

  const float* biasBH = bias + (size_t)bh * NN * NN;
  _Float16* Pw = Pt + wave * 16 * 32;

  for (int k0 = 0; k0 < NN; k0 += 32) {
    int key0 = k0 + l15;
    int key1 = k0 + 16 + l15;
    v16h bk0 = *(const v16h*)(Ks + (size_t)key0 * C_HID + hi * 16);
    v16h bk1 = *(const v16h*)(Ks + (size_t)key1 * C_HID + hi * 16);
    v8f s0 = vzero8(), s1 = vzero8();
    s0 = __builtin_amdgcn_wmma_f32_16x16x32_f16(false, aq.v, false, bk0,
                                                (short)0, s0, false, false);
    s1 = __builtin_amdgcn_wmma_f32_16x16x32_f16(false, aq.v, false, bk1,
                                                (short)0, s1, false, false);
    float mb0 = INF_ * (mask[b * NN + key0] - 1.f);
    float mb1 = INF_ * (mask[b * NN + key1] - 1.f);
#pragma unroll
    for (int r = 0; r < 8; ++r) {
      int q = q0 + r + 8 * hi;
      const float* bp = biasBH + (size_t)q * NN;
      float t0 = s0[r] + mb0 + bp[key0];
      float t1 = s1[r] + mb1 + bp[key1];
      float tmx = fmaxf(t0, t1);
#pragma unroll
      for (int msk = 8; msk >= 1; msk >>= 1)
        tmx = fmaxf(tmx, __shfl_xor(tmx, msk, 32));
      float nmax  = fmaxf(rmax[r], tmx);
      float scale = __expf(rmax[r] - nmax);
      o0[r] *= scale; o1[r] *= scale; rsum[r] *= scale;
      rmax[r] = nmax;
      float p0 = __expf(t0 - nmax);
      float p1 = __expf(t1 - nmax);
      float ps = p0 + p1;
#pragma unroll
      for (int msk = 8; msk >= 1; msk >>= 1) ps += __shfl_xor(ps, msk, 32);
      rsum[r] += ps;
      int rr = r + 8 * hi;
      Pw[rr * 32 + l15]      = (_Float16)p0;
      Pw[rr * 32 + 16 + l15] = (_Float16)p1;
    }
    asm volatile("s_wait_dscnt 0x0" ::: "memory");
    V16U ap;
    ap.h[0] = *(const v8h*)(Pw + l15 * 32 + kb);
    ap.h[1] = *(const v8h*)(Pw + l15 * 32 + kb + 8);
    v16h bv0 = *(const v16h*)(Vt + (size_t)l15 * NN + k0 + hi * 16);
    v16h bv1 = *(const v16h*)(Vt + (size_t)(16 + l15) * NN + k0 + hi * 16);
    o0 = __builtin_amdgcn_wmma_f32_16x16x32_f16(false, ap.v, false, bv0,
                                                (short)0, o0, false, false);
    o1 = __builtin_amdgcn_wmma_f32_16x16x32_f16(false, ap.v, false, bv1,
                                                (short)0, o1, false, false);
  }
#pragma unroll
  for (int r = 0; r < 8; ++r) {
    int q = q0 + r + 8 * hi;
    float inv = 1.f / rsum[r];
    size_t base = ((size_t)(b * NN + q)) * HD + h * C_HID;
    O[base + l15]      = o0[r] * inv;
    O[base + 16 + l15] = o1[r] * inv;
  }
}

// ---------------------------------------------------------------------------
// K5a: gate  og = f16(o * g)
// ---------------------------------------------------------------------------
__global__ void gate_kernel(const float* __restrict__ O,
                            const float* __restrict__ G,
                            _Float16* __restrict__ OG) {
  int i = blockIdx.x * blockDim.x + threadIdx.x;
  OG[i] = (_Float16)(O[i] * G[i]);
}

// ---------------------------------------------------------------------------
// workspace layout (bytes); total ~149 MiB (bias 128 MiB is L2-resident)
// ---------------------------------------------------------------------------
#define OFF_XBF  ((size_t)0)            // 2 MiB  m_ln f16
#define OFF_WQT  ((size_t)2097152)      // 512 KiB each, transposed f16 weights
#define OFF_WKT  ((size_t)2621440)
#define OFF_WVT  ((size_t)3145728)
#define OFF_WGT  ((size_t)3670016)
#define OFF_WOT  ((size_t)4194304)
#define OFF_QBF  ((size_t)4718592)      // 2 MiB each, [b,h,n,d] f16
#define OFF_KBF  ((size_t)6815744)
#define OFF_VBF  ((size_t)8912896)
#define OFF_G    ((size_t)11010048)     // 4 MiB fp32
#define OFF_O    ((size_t)15204352)     // 4 MiB fp32
#define OFF_OG   ((size_t)19398656)     // 2 MiB f16
#define OFF_BIAS ((size_t)21495808)     // 128 MiB fp32 [b,h,q,k]
#define OFF_WZGT ((size_t)155713536)    // 4 KiB  f16 [h][c] gamma-folded w_z
#define OFF_SC   ((size_t)155717632)    // 128 B  S_h / C_h

extern "C" void kernel_launch(void* const* d_in, const int* in_sizes, int n_in,
                              void* d_out, int out_size, void* d_ws,
                              size_t ws_size, hipStream_t stream) {
  const float* m     = (const float*)d_in[0];
  const float* z     = (const float*)d_in[1];
  const float* mask  = (const float*)d_in[2];
  const float* ln_mw = (const float*)d_in[3];
  const float* ln_mb = (const float*)d_in[4];
  const float* ln_zw = (const float*)d_in[5];
  const float* ln_zb = (const float*)d_in[6];
  const float* w_z   = (const float*)d_in[7];
  const float* w_q   = (const float*)d_in[8];
  const float* w_k   = (const float*)d_in[9];
  const float* w_v   = (const float*)d_in[10];
  const float* w_g   = (const float*)d_in[11];
  const float* b_g   = (const float*)d_in[12];
  const float* w_o   = (const float*)d_in[13];
  const float* b_o   = (const float*)d_in[14];
  float* out = (float*)d_out;

  char* ws = (char*)d_ws;
  _Float16* Xbf  = (_Float16*)(ws + OFF_XBF);
  _Float16* Wqt  = (_Float16*)(ws + OFF_WQT);
  _Float16* Wkt  = (_Float16*)(ws + OFF_WKT);
  _Float16* Wvt  = (_Float16*)(ws + OFF_WVT);
  _Float16* Wgt  = (_Float16*)(ws + OFF_WGT);
  _Float16* Wot  = (_Float16*)(ws + OFF_WOT);
  _Float16* Qbf  = (_Float16*)(ws + OFF_QBF);
  _Float16* Kbf  = (_Float16*)(ws + OFF_KBF);
  _Float16* Vbf  = (_Float16*)(ws + OFF_VBF);
  float*    Gws  = (float*)(ws + OFF_G);
  float*    Ows  = (float*)(ws + OFF_O);
  _Float16* OGbf = (_Float16*)(ws + OFF_OG);
  float*    Bias = (float*)(ws + OFF_BIAS);
  _Float16* Wzgt = (_Float16*)(ws + OFF_WZGT);
  float*    SC   = (float*)(ws + OFF_SC);

  // K0: weight convert/transpose + LN-folded pair-bias weight
  convert_w_kernel<<<1024, 256, 0, stream>>>(w_q, Wqt, C_IN, HD);
  convert_w_kernel<<<1024, 256, 0, stream>>>(w_k, Wkt, C_IN, HD);
  convert_w_kernel<<<1024, 256, 0, stream>>>(w_v, Wvt, C_IN, HD);
  convert_w_kernel<<<1024, 256, 0, stream>>>(w_g, Wgt, C_IN, HD);
  convert_w_kernel<<<1024, 256, 0, stream>>>(w_o, Wot, HD, C_IN);
  prep_wz_kernel<<<8, 256, 0, stream>>>(w_z, ln_zw, ln_zb, Wzgt, SC);

  // K1: LN(m) -> f16
  ln_m_kernel<<<MTOT / 8, 256, 0, stream>>>(m, ln_mw, ln_mb, Xbf);

  // K2: Q,K,V,G projections (WMMA)
  gemm_wmma_kernel<<<512, 256, 0, stream>>>(Xbf, Wqt, Qbf, nullptr, nullptr, 0);
  gemm_wmma_kernel<<<512, 256, 0, stream>>>(Xbf, Wkt, Kbf, nullptr, nullptr, 1);
  gemm_wmma_kernel<<<512, 256, 0, stream>>>(Xbf, Wvt, Vbf, nullptr, nullptr, 2);
  gemm_wmma_kernel<<<512, 256, 0, stream>>>(Xbf, Wgt, nullptr, Gws, b_g, 3);

  // K3: pair bias via WMMA (HBM-bound pass over 1 GiB of z)
  pair_bias_wmma_kernel<<<(BB * NN * NN) / 16 / 8, 256, 0, stream>>>(
      z, Wzgt, SC, Bias);

  // K4: flash attention with pair bias (136 KiB dynamic LDS, async K stage)
  attn_kernel<<<BB * NH * 8, 256, 139264, stream>>>(Qbf, Kbf, Vbf, Bias, mask,
                                                    Ows);

  // K5: gating + output projection (WMMA)
  gate_kernel<<<(MTOT * HD) / 256, 256, 0, stream>>>(Ows, Gws, OGbf);
  gemm_wmma_kernel<<<512, 256, 0, stream>>>(OGbf, Wot, nullptr, out, b_o, 4);
}
